// VectorQuantizer2_38835094290487
// MI455X (gfx1250) — matrix operand order repaired and checked by reading, not compiled, not run on hardware
//
#include <hip/hip_runtime.h>
#include <hip/hip_bf16.h>

typedef __attribute__((ext_vector_type(16))) _Float16 v16h;
typedef __attribute__((ext_vector_type(8)))  float    v8f;

#define Cc 32
#define Ll 1024
#define Bb 32

// ---------------------------------------------------------------------------
// init: f_rest = f, f_hat = 0, zero loss slots
// ---------------------------------------------------------------------------
__global__ void k_init(const float* __restrict__ f, float* __restrict__ f_rest,
                       float* __restrict__ f_hat, float* __restrict__ loss_slots, int n) {
    int i = blockIdx.x * blockDim.x + threadIdx.x;
    if (i < n) { f_rest[i] = f[i]; f_hat[i] = 0.0f; }
    if (blockIdx.x == 0 && threadIdx.x < 8) loss_slots[threadIdx.x] = 0.0f;
}

// ---------------------------------------------------------------------------
// codebook prep: emb (f32) -> emb16 (f16), e_sq[v] = ||emb[v]||^2
// ---------------------------------------------------------------------------
__global__ void k_prep_emb(const float* __restrict__ emb, _Float16* __restrict__ emb16,
                           float* __restrict__ e_sq, int V) {
    int v = blockIdx.x * blockDim.x + threadIdx.x;
    if (v >= V) return;
    const float* src = emb + (size_t)v * Cc;
    _Float16* dst = emb16 + (size_t)v * Cc;
    float s = 0.0f;
#pragma unroll
    for (int c = 0; c < Cc; ++c) {
        float x = src[c];
        s = fmaf(x, x, s);
        dst[c] = (_Float16)x;
    }
    e_sq[v] = s;
}

// ---------------------------------------------------------------------------
// downsample f_rest [B,C,L] -> z16 [(B*pl), C] (linear interp, align_corners=F)
// ---------------------------------------------------------------------------
__global__ void k_downsample(const float* __restrict__ f_rest, _Float16* __restrict__ z16,
                             int pl) {
    int i = blockIdx.x * blockDim.x + threadIdx.x;
    int n = Bb * pl * Cc;
    if (i >= n) return;
    int c = i & (Cc - 1);
    int row = i >> 5;            // b*pl + p
    int p = row % pl;
    int b = row / pl;
    float ratio = (float)Ll / (float)pl;
    float pos = (p + 0.5f) * ratio - 0.5f;
    pos = fminf(fmaxf(pos, 0.0f), (float)(Ll - 1));
    int lo = (int)floorf(pos);
    int hi = min(lo + 1, Ll - 1);
    float w = pos - (float)lo;
    const float* src = f_rest + ((size_t)b * Cc + c) * Ll;
    float v = src[lo] * (1.0f - w) + src[hi] * w;
    z16[(size_t)row * Cc + c] = (_Float16)v;
}

// ---------------------------------------------------------------------------
// argmin over codebook via V_WMMA_F32_16X16X32_F16, 2x software pipelined.
// One block = 16 rows of z. 8 waves; each wave handles column-tile pairs
// (ct, ct+8) with stride 16 so two independent WMMA accumulators are in
// flight -> the 4-slot WMMA->VALU hazard window is filled with real work.
// d = e_sq[code] - 2*dot(z,e)   (||z||^2 is row-constant -> dropped)
// ---------------------------------------------------------------------------
__global__ void __launch_bounds__(256)
k_argmin_wmma(const _Float16* __restrict__ z16, const _Float16* __restrict__ emb16,
              const float* __restrict__ e_sq, int* __restrict__ idx_out, int V) {
    __shared__ float smin[8][16];
    __shared__ int   sidx[8][16];

    int t    = threadIdx.x;
    int wave = t >> 5;
    int lane = t & 31;
    int half = lane >> 4;   // 0: lanes 0-15, 1: lanes 16-31
    int ln   = lane & 15;
    int tile = blockIdx.x;

    // --- A fragment: 16x32 f16, row = tile*16 + ln ---
    // lane<16 : K = {0..15}; lane>=16 : K = {16..31}
    const _Float16* zr = z16 + ((size_t)(tile * 16 + ln)) * Cc + half * 16;
    v16h a;
#pragma unroll
    for (int i = 0; i < 8; ++i) a[i] = zr[i];
#pragma unroll
    for (int i = 0; i < 8; ++i) a[8 + i] = zr[8 + i];

    float best[8];
    int   bidx[8];
#pragma unroll
    for (int s = 0; s < 8; ++s) { best[s] = 3.4e38f; bidx[s] = 0; }

    const int nct = V >> 4;
    const size_t bstride = (size_t)half * 16;   // K-half select within a code row

    int ct = wave;
    for (; ct + 8 < nct; ct += 16) {
        int ct0 = ct, ct1 = ct + 8;
        const _Float16* ec0 = emb16 + ((size_t)(ct0 * 16 + ln)) * Cc + bstride;
        const _Float16* ec1 = emb16 + ((size_t)(ct1 * 16 + ln)) * Cc + bstride;
        v16h bf0, bf1;
#pragma unroll
        for (int i = 0; i < 16; ++i) bf0[i] = ec0[i];
#pragma unroll
        for (int i = 0; i < 16; ++i) bf1[i] = ec1[i];

        int code0 = ct0 * 16 + ln;
        int code1 = ct1 * 16 + ln;
        float es0 = e_sq[code0];
        float es1 = e_sq[code1];

        v8f acc0 = {};
        v8f acc1 = {};
        acc0 = __builtin_amdgcn_wmma_f32_16x16x32_f16(
            false, a, false, bf0, (short)0, acc0, false, false);
        acc1 = __builtin_amdgcn_wmma_f32_16x16x32_f16(
            false, a, false, bf1, (short)0, acc1, false, false);

#pragma unroll
        for (int s = 0; s < 8; ++s) {
            float d0 = fmaf(-2.0f, acc0[s], es0);
            if (d0 < best[s]) { best[s] = d0; bidx[s] = code0; }
        }
#pragma unroll
        for (int s = 0; s < 8; ++s) {
            float d1 = fmaf(-2.0f, acc1[s], es1);
            if (d1 < best[s]) { best[s] = d1; bidx[s] = code1; }
        }
    }
    // tail (not taken for V=4096, kept for generality)
    for (; ct < nct; ct += 8) {
        const _Float16* ec = emb16 + ((size_t)(ct * 16 + ln)) * Cc + bstride;
        v16h bf;
#pragma unroll
        for (int i = 0; i < 16; ++i) bf[i] = ec[i];
        v8f acc = {};
        acc = __builtin_amdgcn_wmma_f32_16x16x32_f16(
            false, a, false, bf, (short)0, acc, false, false);
        int code = ct * 16 + ln;
        float es = e_sq[code];
#pragma unroll
        for (int s = 0; s < 8; ++s) {
            float d = fmaf(-2.0f, acc[s], es);
            if (d < best[s]) { best[s] = d; bidx[s] = code; }
        }
    }

    // reduce across the 16 lanes of each half (columns)
#pragma unroll
    for (int m = 1; m < 16; m <<= 1) {
#pragma unroll
        for (int s = 0; s < 8; ++s) {
            float ov = __shfl_xor(best[s], m);
            int   oi = __shfl_xor(bidx[s], m);
            if (ov < best[s] || (ov == best[s] && oi < bidx[s])) {
                best[s] = ov; bidx[s] = oi;
            }
        }
    }
    if (ln == 0) {
#pragma unroll
        for (int s = 0; s < 8; ++s) {
            smin[wave][s + 8 * half] = best[s];
            sidx[wave][s + 8 * half] = bidx[s];
        }
    }
    __syncthreads();

    // reduce across the 8 waves
    if (t < 16) {
        float bm = smin[0][t]; int bi = sidx[0][t];
#pragma unroll
        for (int w = 1; w < 8; ++w) {
            float ov = smin[w][t]; int oi = sidx[w][t];
            if (ov < bm || (ov == bm && oi < bi)) { bm = ov; bi = oi; }
        }
        idx_out[tile * 16 + t] = bi;
    }
}

// ---------------------------------------------------------------------------
// gather codes + upsample pl -> L:  h_up[b][c][l]
// ---------------------------------------------------------------------------
__global__ void k_gather_upsample(const float* __restrict__ emb, const int* __restrict__ idx,
                                  float* __restrict__ h_up, int pl) {
    int i = blockIdx.x * blockDim.x + threadIdx.x;
    if (i >= Bb * Cc * Ll) return;
    int l = i & (Ll - 1);
    int c = (i >> 10) & (Cc - 1);
    int b = i >> 15;
    float ratio = (float)pl / (float)Ll;
    float pos = (l + 0.5f) * ratio - 0.5f;
    pos = fminf(fmaxf(pos, 0.0f), (float)(pl - 1));
    int lo = (int)floorf(pos);
    int hi = min(lo + 1, pl - 1);
    float w = pos - (float)lo;
    int i0 = idx[b * pl + lo];
    int i1 = idx[b * pl + hi];
    float h = emb[(size_t)i0 * Cc + c] * (1.0f - w) + emb[(size_t)i1 * Cc + c] * w;
    h_up[i] = h;
}

// ---------------------------------------------------------------------------
// Phi conv (C=32->32, ks=3, pad=1) + residual/accumulator update + loss partial
// one block: (b, 64 output positions). LDS: 32x66 halo tile + weights.
// h_final = 0.5*h_up + 0.5*(conv+bias); f_hat += h; f_rest -= h;
// loss_slot += sum((f_hat_new - f)^2)
// ---------------------------------------------------------------------------
__global__ void __launch_bounds__(256)
k_conv_update(const float* __restrict__ h_up, const float* __restrict__ w3,
              const float* __restrict__ bias, const float* __restrict__ f,
              float* __restrict__ f_hat, float* __restrict__ f_rest,
              float* __restrict__ loss_slot) {
    __shared__ float sh[Cc * 66];
    __shared__ float wsh[Cc * Cc * 3];
    __shared__ float bsh[Cc];
    __shared__ float red[8];

    const int nblkL = Ll / 64;
    int b  = blockIdx.x / nblkL;
    int l0 = (blockIdx.x % nblkL) * 64;
    int t  = threadIdx.x;

    for (int e = t; e < Cc * 66; e += 256) {
        int ci = e / 66, li = e - ci * 66;
        int l = l0 + li - 1;
        sh[e] = (l >= 0 && l < Ll) ? h_up[((size_t)b * Cc + ci) * Ll + l] : 0.0f;
    }
    for (int e = t; e < Cc * Cc * 3; e += 256) wsh[e] = w3[e];
    if (t < Cc) bsh[t] = bias[t];
    __syncthreads();

    int co = t & (Cc - 1);
    float local = 0.0f;
    for (int li = t >> 5; li < 64; li += 8) {
        float y = bsh[co];
#pragma unroll
        for (int ci = 0; ci < Cc; ++ci) {
            const float* wp = wsh + (co * Cc + ci) * 3;
            const float* sp = sh + ci * 66 + li;
            y = fmaf(wp[0], sp[0], y);
            y = fmaf(wp[1], sp[1], y);
            y = fmaf(wp[2], sp[2], y);
        }
        float hf = 0.5f * sh[co * 66 + li + 1] + 0.5f * y;
        size_t off = ((size_t)b * Cc + co) * Ll + (l0 + li);
        float fh = f_hat[off] + hf;
        f_hat[off] = fh;
        f_rest[off] -= hf;
        float diff = fh - f[off];
        local = fmaf(diff, diff, local);
    }

#pragma unroll
    for (int m = 1; m < 32; m <<= 1) local += __shfl_xor(local, m);
    if ((t & 31) == 0) red[t >> 5] = local;
    __syncthreads();
    if (t == 0) {
        float s = 0.0f;
#pragma unroll
        for (int w = 0; w < 8; ++w) s += red[w];
        atomicAdd(loss_slot, s);
    }
}

// ---------------------------------------------------------------------------
// finalize: out[0..BCL) = f_hat (STE output == f_hat numerically), out[BCL]=loss
// ---------------------------------------------------------------------------
__global__ void k_finalize(const float* __restrict__ f_hat, const float* __restrict__ loss_slots,
                           float* __restrict__ out, int n) {
    int i = blockIdx.x * blockDim.x + threadIdx.x;
    if (i < n) out[i] = f_hat[i];
    if (i == 0) {
        float s = 0.0f;
#pragma unroll
        for (int si = 0; si < 6; ++si) s += 1.25f * loss_slots[si];
        out[n] = s / (float)n / 6.0f;
    }
}

// ---------------------------------------------------------------------------
extern "C" void kernel_launch(void* const* d_in, const int* in_sizes, int n_in,
                              void* d_out, int out_size, void* d_ws, size_t ws_size,
                              hipStream_t stream) {
    const float* f     = (const float*)d_in[0];   // [32,32,1024]
    const float* emb   = (const float*)d_in[1];   // [V,32]
    const float* phi_w = (const float*)d_in[2];   // [4,32,32,3]
    const float* phi_b = (const float*)d_in[3];   // [4,32]

    const int BCL = in_sizes[0];                  // 1048576
    const int V   = in_sizes[1] / Cc;             // 4096
    const int max_rows = Bb * Ll;                 // 32768

    // workspace layout (256B aligned slices)
    char* ws = (char*)d_ws;
    size_t off = 0;
    auto carve = [&](size_t bytes) { void* p = ws + off; off = (off + bytes + 255) & ~(size_t)255; return p; };
    float*    f_rest     = (float*)   carve((size_t)BCL * 4);
    float*    f_hat      = (float*)   carve((size_t)BCL * 4);
    float*    h_up       = (float*)   carve((size_t)BCL * 4);
    _Float16* z16        = (_Float16*)carve((size_t)max_rows * Cc * 2);
    _Float16* emb16      = (_Float16*)carve((size_t)V * Cc * 2);
    float*    e_sq       = (float*)   carve((size_t)V * 4);
    int*      idx        = (int*)     carve((size_t)max_rows * 4);
    float*    loss_slots = (float*)   carve(8 * 4);
    (void)ws_size; (void)n_in; (void)out_size;

    const int pls[6]  = {1, 4, 16, 64, 256, 1024};
    const int pidx[6] = {0, 0, 1, 2, 3, 3};      // PhiPartiallyShared tick lookup, K=4

    k_init<<<(BCL + 255) / 256, 256, 0, stream>>>(f, f_rest, f_hat, loss_slots, BCL);
    k_prep_emb<<<(V + 255) / 256, 256, 0, stream>>>(emb, emb16, e_sq, V);

    for (int si = 0; si < 6; ++si) {
        const int pl = pls[si];
        const int nrows = Bb * pl;
        const int nz = nrows * Cc;

        k_downsample<<<(nz + 255) / 256, 256, 0, stream>>>(f_rest, z16, pl);
        k_argmin_wmma<<<nrows / 16, 256, 0, stream>>>(z16, emb16, e_sq, idx, V);
        k_gather_upsample<<<(BCL + 255) / 256, 256, 0, stream>>>(emb, idx, h_up, pl);
        k_conv_update<<<Bb * (Ll / 64), 256, 0, stream>>>(
            h_up, phi_w + (size_t)pidx[si] * Cc * Cc * 3, phi_b + (size_t)pidx[si] * Cc,
            f, f_hat, f_rest, loss_slots + si);
    }

    k_finalize<<<(BCL + 255) / 256, 256, 0, stream>>>(f_hat, loss_slots, (float*)d_out, BCL);
}
